// MLA_42116449305263
// MI455X (gfx1250) — matrix operand order repaired and checked by reading, not compiled, not run on hardware
//
#include <hip/hip_runtime.h>
#include <hip/hip_bf16.h>

typedef __attribute__((ext_vector_type(16))) _Float16 v16h;
typedef __attribute__((ext_vector_type(8)))  float    v8f;
typedef __attribute__((ext_vector_type(4)))  unsigned int u32x4;
typedef __attribute__((ext_vector_type(8)))  int      i32x8;
typedef __attribute__((ext_vector_type(4)))  int      i32x4;

#define T_SEQ   4096
#define D_MODEL 2048
#define H_HEADS 16
#define DN      128     // D_NOPE
#define DR      64      // D_ROPE
#define DV      128     // D_V
#define DQK     192
#define KVR     512
#define QSTRIDE (H_HEADS*DQK)     // 3072
#define KVBSTR  (H_HEADS*(DN+DV)) // 4096
#define OSTRIDE (H_HEADS*DV)      // 2048
#define SCALE   0.07216878364870322f  // 192^-0.5
#define LN10000 9.210340371976184f

// ---------------- WMMA helpers ----------------

__device__ inline v8f wmma16(v16h a, v16h b, v8f c) {
  return __builtin_amdgcn_wmma_f32_16x16x32_f16(
      /*neg_a=*/false, a, /*neg_b=*/false, b,
      /*c_mod=*/(short)0, c, /*reuse_a=*/false, /*reuse_b=*/false);
}

// A-fragment (16x32 f16, MxK): lane l -> row M=l%16, element j -> K = j + 16*(l/16).
// p must already point at row base + half*16; 16 contiguous f16.
__device__ inline v16h load_afrag(const _Float16* p) {
  v16h f;
#pragma unroll
  for (int j = 0; j < 16; ++j) f[j] = p[j];
  return f;
}

// B-fragment (32x16 f16, KxN): lane l -> col N=l%16, element j -> K = j + 16*(l/16).
// p must already point at col's K-major base + half*16; 16 contiguous f16.
__device__ inline v16h load_bfrag(const _Float16* p) {
  v16h f;
#pragma unroll
  for (int j = 0; j < 16; ++j) f[j] = p[j];
  return f;
}

__device__ inline float red_max16(float v) {
#pragma unroll
  for (int off = 1; off < 16; off <<= 1) v = fmaxf(v, __shfl_xor(v, off, 32));
  return v;
}
__device__ inline float red_sum16(float v) {
#pragma unroll
  for (int off = 1; off < 16; off <<= 1) v += __shfl_xor(v, off, 32);
  return v;
}

// ---------------- Tensor Data Mover: 2D tile -> LDS ----------------
// Loads a [128 x 32] f16 tile (row stride = row_stride_elems in memory) into LDS
// with hardware row padding 64B->80B (pad_interval=16 DW, pad_amount=4 DW), i.e.
// the 40-f16 padded pitch used by the WMMA fragment readers. Rows past
// rows_remaining and cols past k_remaining zero-fill via D# OOB semantics.
__device__ inline void tdm_load_tile(const _Float16* gbase, unsigned lds_addr,
                                     int rows_remaining, int row_stride_elems,
                                     int k_remaining) {
  unsigned long long ga = (unsigned long long)(size_t)gbase;
  u32x4 g0;
  g0[0] = 1u;                                   // count=1 (valid user descriptor)
  g0[1] = lds_addr;                             // LDS byte address
  g0[2] = (unsigned)(ga & 0xffffffffu);         // global_addr[31:0]
  g0[3] = (unsigned)((ga >> 32) & 0x01ffffffu)  // global_addr[56:32]
          | 0x80000000u;                        // type=2 ("image")
  unsigned td0 = (unsigned)(k_remaining  < 0 ? 0 : k_remaining);
  unsigned td1 = (unsigned)(rows_remaining < 0 ? 0 : rows_remaining);
  unsigned st  = (unsigned)row_stride_elems;
  i32x8 g1;
  g1[0] = (int)((1u << 16)      // data_size = 2 bytes
              | (1u << 20)      // pad_enable
              | (3u << 22)      // pad_interval: 16 DWORDs (one 64B row)
              | (3u << 25));    // pad_amount: 4 DWORDs (16B -> 80B pitch)
  g1[1] = (int)((td0 & 0xffffu) << 16);              // tensor_dim0[15:0]
  g1[2] = (int)((td0 >> 16) | ((td1 & 0xffffu) << 16)); // td0 hi | td1 lo
  g1[3] = (int)((td1 >> 16) | (32u << 16));          // td1 hi | tile_dim0=32
  g1[4] = (int)128;                                  // tile_dim1=128, tile_dim2=0
  g1[5] = (int)st;                                   // tensor_dim0_stride[31:0]
  g1[6] = 0;                                         // stride hi | dim1_stride lo
  g1[7] = 0;
  i32x4 z4 = {0, 0, 0, 0};
  i32x8 z8 = {0, 0, 0, 0, 0, 0, 0, 0};
  __builtin_amdgcn_tensor_load_to_lds(g0, g1, z4, z4, z8, 0);
}

// ---------------- generic f16 WMMA GEMM: C[M,N] = A[M,K] * B[N,K]^T ----------------
// 128x128 C tile per block, 8 waves (4x2), each wave 32x64 via 2x4 16x16 frags,
// K-step 32. Double-buffered LDS fed by the Tensor Data Mover (wave 0 issues,
// TENSORcnt + block barrier synchronize).
__global__ __launch_bounds__(256) void gemm_f16_wmma(
    const _Float16* __restrict__ A, const _Float16* __restrict__ B,
    float* __restrict__ Cf, _Float16* __restrict__ Ch,
    int M, int N, int K) {
  __shared__ _Float16 As[2][128 * 40];
  __shared__ _Float16 Bs[2][128 * 40];
  const int tid  = threadIdx.x;
  const int wave = tid >> 5, lane = tid & 31;
  const int l16  = lane & 15, half = lane >> 4;
  const int bm = blockIdx.x * 128, bn = blockIdx.y * 128;
  const int wm = (wave & 3) * 32, wn = (wave >> 2) * 64;

  v8f acc[2][4];
#pragma unroll
  for (int i = 0; i < 2; ++i)
#pragma unroll
    for (int j = 0; j < 4; ++j) acc[i][j] = 0.0f;

  const unsigned ldsA[2] = {(unsigned)(size_t)&As[0][0], (unsigned)(size_t)&As[1][0]};
  const unsigned ldsB[2] = {(unsigned)(size_t)&Bs[0][0], (unsigned)(size_t)&Bs[1][0]};

  // prologue: stream first tiles into buffer 0
  if (wave == 0) {
    tdm_load_tile(A + (size_t)bm * K, ldsA[0], M - bm, K, K);
    tdm_load_tile(B + (size_t)bn * K, ldsB[0], N - bn, K, K);
  }

  int cur = 0;
  for (int k0 = 0; k0 < K; k0 += 32) {
    if (wave == 0) {
      if (k0 + 32 < K) {
        // prefetch next tiles into the other buffer (its readers finished at
        // the barrier that ended the previous iteration)
        tdm_load_tile(A + (size_t)bm * K + (k0 + 32), ldsA[cur ^ 1], M - bm, K, K - (k0 + 32));
        tdm_load_tile(B + (size_t)bn * K + (k0 + 32), ldsB[cur ^ 1], N - bn, K, K - (k0 + 32));
        __builtin_amdgcn_s_wait_tensorcnt(2);  // current buffer's pair complete
      } else {
        __builtin_amdgcn_s_wait_tensorcnt(0);
      }
    }
    __syncthreads();

    const _Float16* as = &As[cur][0];
    const _Float16* bs = &Bs[cur][0];
    v16h af[2], bf[4];
#pragma unroll
    for (int i = 0; i < 2; ++i)
      af[i] = load_afrag(&as[(wm + i * 16 + l16) * 40 + half * 16]);
#pragma unroll
    for (int j = 0; j < 4; ++j)
      bf[j] = load_bfrag(&bs[(wn + j * 16 + l16) * 40 + half * 16]);

#pragma unroll
    for (int i = 0; i < 2; ++i)
#pragma unroll
      for (int j = 0; j < 4; ++j)
        acc[i][j] = wmma16(af[i], bf[j], acc[i][j]);
    __syncthreads();
    cur ^= 1;
  }

  // store: C element (r,lane) -> M = r + 8*half, N = lane%16
#pragma unroll
  for (int i = 0; i < 2; ++i)
#pragma unroll
    for (int j = 0; j < 4; ++j) {
      int n = bn + wn + j * 16 + l16;
#pragma unroll
      for (int r = 0; r < 8; ++r) {
        int mm = bm + wm + i * 16 + r + 8 * half;
        if (mm < M && n < N) {
          float v = acc[i][j][r];
          if (Ch) Ch[(size_t)mm * N + n] = (_Float16)v;
          else    Cf[(size_t)mm * N + n] = v;
        }
      }
    }
}

// ---------------- elementwise kernels ----------------

__global__ void f32_to_f16(const float* __restrict__ in, _Float16* __restrict__ out, size_t n) {
  size_t i = (size_t)blockIdx.x * blockDim.x + threadIdx.x;
  if (i < n) out[i] = (_Float16)in[i];
}

// RoPE on q_pe (in-place on q f16 [T][H*192], pe at +128 per head)
__global__ void rope_q_kernel(_Float16* __restrict__ q) {
  size_t i = (size_t)blockIdx.x * blockDim.x + threadIdx.x;
  if (i >= (size_t)T_SEQ * H_HEADS * 32) return;
  int p = (int)(i & 31);
  int h = (int)((i >> 5) & 15);
  int t = (int)(i >> 9);
  float inv_freq = __expf(-((float)(2 * p) / 64.0f) * LN10000);
  float ang = (float)t * inv_freq;
  float s, c;
  __sincosf(ang, &s, &c);
  _Float16* ptr = q + (size_t)t * QSTRIDE + h * DQK + DN + 2 * p;
  float x1 = (float)ptr[0], x2 = (float)ptr[1];
  ptr[0] = (_Float16)(x1 * c - x2 * s);
  ptr[1] = (_Float16)(x1 * s + x2 * c);
}

// LayerNorm over KVR latent + RoPE on k_pe. One block per token.
__global__ __launch_bounds__(256) void ln_ropek_kernel(
    const float* __restrict__ kvfull, const float* __restrict__ gamma,
    const float* __restrict__ beta, _Float16* __restrict__ kvln,
    _Float16* __restrict__ kpe) {
  int t = blockIdx.x, tid = threadIdx.x;
  int wave = tid >> 5, lane = tid & 31;
  const float* row = kvfull + (size_t)t * (KVR + DR);
  float s = 0.f, s2 = 0.f;
  for (int i = tid; i < KVR; i += 256) { float v = row[i]; s += v; s2 += v * v; }
#pragma unroll
  for (int off = 1; off < 32; off <<= 1) {
    s  += __shfl_xor(s, off, 32);
    s2 += __shfl_xor(s2, off, 32);
  }
  __shared__ float red[18];
  if (lane == 0) { red[wave] = s; red[8 + wave] = s2; }
  __syncthreads();
  if (tid == 0) {
    float a = 0.f, b = 0.f;
    for (int w = 0; w < 8; ++w) { a += red[w]; b += red[8 + w]; }
    float mu = a / (float)KVR;
    float var = b / (float)KVR - mu * mu;
    red[16] = mu;
    red[17] = rsqrtf(var + 1e-5f);
  }
  __syncthreads();
  float mu = red[16], rstd = red[17];
  for (int i = tid; i < KVR; i += 256)
    kvln[(size_t)t * KVR + i] = (_Float16)((row[i] - mu) * rstd * gamma[i] + beta[i]);
  if (tid < 32) {
    int p = tid;
    float inv_freq = __expf(-((float)(2 * p) / 64.0f) * LN10000);
    float ang = (float)t * inv_freq;
    float sn, cs;
    __sincosf(ang, &sn, &cs);
    float x1 = row[KVR + 2 * p], x2 = row[KVR + 2 * p + 1];
    kpe[(size_t)t * DR + 2 * p]     = (_Float16)(x1 * cs - x2 * sn);
    kpe[(size_t)t * DR + 2 * p + 1] = (_Float16)(x1 * sn + x2 * cs);
  }
}

// vT[(h*128+d)][t] = kvb[t][h*256 + 128 + d]  (dv-major V for contiguous B-fragments)
__global__ void transpose_v_kernel(const _Float16* __restrict__ kvb, _Float16* __restrict__ vT) {
  size_t i = (size_t)blockIdx.x * blockDim.x + threadIdx.x;
  if (i >= (size_t)H_HEADS * DV * T_SEQ) return;
  int t = (int)(i & (T_SEQ - 1));
  size_t hd = i >> 12;
  int d = (int)(hd & (DV - 1));
  int h = (int)(hd >> 7);
  vT[i] = kvb[(size_t)t * KVBSTR + h * (DN + DV) + DN + d];
}

// ---------------- flash attention (per head, 128 q rows per block, 16 per wave) ----------------
__global__ __launch_bounds__(256) void mla_attn_kernel(
    const _Float16* __restrict__ q,    // [T][3072]
    const _Float16* __restrict__ kvb,  // [T][4096] (k_nope at h*256)
    const _Float16* __restrict__ kpe,  // [T][64]
    const _Float16* __restrict__ vT,   // [H*128][T]
    _Float16* __restrict__ o)          // [T][2048]
{
  const int h    = blockIdx.y;
  const int wave = threadIdx.x >> 5, lane = threadIdx.x & 31;
  const int l16  = lane & 15, half = lane >> 4;
  const int q0   = blockIdx.x * 128 + wave * 16;
  const int qhi  = q0 + 15;

  __shared__ _Float16 Plds[8 * 16 * 40];
  _Float16* pl = &Plds[wave * 16 * 40];

  // Q fragments (persist across the key loop)
  const _Float16* qrow = q + (size_t)(q0 + l16) * QSTRIDE + h * DQK;
  v16h qn[4], qpf[2];
#pragma unroll
  for (int c = 0; c < 4; ++c) qn[c] = load_afrag(qrow + c * 32 + half * 16);
#pragma unroll
  for (int c = 0; c < 2; ++c) qpf[c] = load_afrag(qrow + DN + c * 32 + half * 16);

  v8f oacc[8];
#pragma unroll
  for (int nf = 0; nf < 8; ++nf) oacc[nf] = 0.0f;
  float mrow[8], lrow[8];
#pragma unroll
  for (int r = 0; r < 8; ++r) { mrow[r] = -1e30f; lrow[r] = 0.0f; }

  for (int kb = 0; kb <= qhi; kb += 32) {
    v8f s0 = 0.0f, s1 = 0.0f;
#pragma unroll
    for (int c = 0; c < 4; ++c) {
      v16h k0f = load_bfrag(kvb + (size_t)(kb + l16) * KVBSTR + h * (DN + DV) + c * 32 + half * 16);
      s0 = wmma16(qn[c], k0f, s0);
      v16h k1f = load_bfrag(kvb + (size_t)(kb + 16 + l16) * KVBSTR + h * (DN + DV) + c * 32 + half * 16);
      s1 = wmma16(qn[c], k1f, s1);
    }
#pragma unroll
    for (int c = 0; c < 2; ++c) {
      v16h k0f = load_bfrag(kpe + (size_t)(kb + l16) * DR + c * 32 + half * 16);
      s0 = wmma16(qpf[c], k0f, s0);
      v16h k1f = load_bfrag(kpe + (size_t)(kb + 16 + l16) * DR + c * 32 + half * 16);
      s1 = wmma16(qpf[c], k1f, s1);
    }

    float alpha[8];
#pragma unroll
    for (int r = 0; r < 8; ++r) {
      int qr = q0 + r + 8 * half;
      float a = (kb + l16      <= qr) ? s0[r] * SCALE : -1e30f;
      float b = (kb + 16 + l16 <= qr) ? s1[r] * SCALE : -1e30f;
      float mx = red_max16(fmaxf(a, b));
      float mn = fmaxf(mrow[r], mx);
      float al = __expf(mrow[r] - mn);
      float p0 = __expf(a - mn);
      float p1 = __expf(b - mn);
      lrow[r] = lrow[r] * al + red_sum16(p0 + p1);
      mrow[r] = mn;
      alpha[r] = al;
      // transpose P through per-wave LDS strip: [row][key] padded to 40
      pl[(r + 8 * half) * 40 + l16]      = (_Float16)p0;
      pl[(r + 8 * half) * 40 + 16 + l16] = (_Float16)p1;
    }
#pragma unroll
    for (int nf = 0; nf < 8; ++nf)
#pragma unroll
      for (int r = 0; r < 8; ++r) oacc[nf][r] *= alpha[r];

    v16h pf = load_afrag(pl + l16 * 40 + half * 16);
#pragma unroll
    for (int nf = 0; nf < 8; ++nf) {
      v16h vf = load_bfrag(vT + ((size_t)h * DV + nf * 16 + l16) * T_SEQ + kb + half * 16);
      oacc[nf] = wmma16(pf, vf, oacc[nf]);
    }
  }

  float inv[8];
#pragma unroll
  for (int r = 0; r < 8; ++r) inv[r] = 1.0f / lrow[r];
#pragma unroll
  for (int nf = 0; nf < 8; ++nf)
#pragma unroll
    for (int r = 0; r < 8; ++r)
      o[(size_t)(q0 + r + 8 * half) * OSTRIDE + h * DV + nf * 16 + l16] =
          (_Float16)(oacc[nf][r] * inv[r]);
}

// ---------------- host launcher ----------------

extern "C" void kernel_launch(void* const* d_in, const int* in_sizes, int n_in,
                              void* d_out, int out_size, void* d_ws, size_t ws_size,
                              hipStream_t stream) {
  const float* x        = (const float*)d_in[0];
  const float* wq       = (const float*)d_in[1];
  const float* wkv_a    = (const float*)d_in[2];
  const float* kv_gamma = (const float*)d_in[3];
  const float* kv_beta  = (const float*)d_in[4];
  const float* wkv_b    = (const float*)d_in[5];
  const float* wo       = (const float*)d_in[6];
  float* out = (float*)d_out;

  char* ws = (char*)d_ws;
  size_t off = 0;
  auto alloc = [&](size_t bytes) -> char* {
    char* p = ws + off;
    off += (bytes + 255) & ~(size_t)255;
    return p;
  };

  const size_t nX   = (size_t)T_SEQ * D_MODEL;
  const size_t nWQ  = (size_t)QSTRIDE * D_MODEL;
  const size_t nWA  = (size_t)(KVR + DR) * D_MODEL;
  const size_t nWB  = (size_t)KVBSTR * KVR;
  const size_t nWO  = (size_t)D_MODEL * OSTRIDE;

  _Float16* xh     = (_Float16*)alloc(nX * 2);
  _Float16* wqh    = (_Float16*)alloc(nWQ * 2);
  _Float16* wkvah  = (_Float16*)alloc(nWA * 2);
  _Float16* wkvbh  = (_Float16*)alloc(nWB * 2);
  _Float16* woh    = (_Float16*)alloc(nWO * 2);
  _Float16* qbuf   = (_Float16*)alloc((size_t)T_SEQ * QSTRIDE * 2);
  float*    kvfull = (float*)   alloc((size_t)T_SEQ * (KVR + DR) * 4);
  _Float16* kvlnh  = (_Float16*)alloc((size_t)T_SEQ * KVR * 2);
  _Float16* kph    = (_Float16*)alloc((size_t)T_SEQ * DR * 2);
  _Float16* kvbh   = (_Float16*)alloc((size_t)T_SEQ * KVBSTR * 2);
  _Float16* vTh    = (_Float16*)alloc((size_t)H_HEADS * DV * T_SEQ * 2);
  _Float16* attnh  = (_Float16*)alloc((size_t)T_SEQ * OSTRIDE * 2);

  auto cvt = [&](const float* src, _Float16* dst, size_t n) {
    f32_to_f16<<<dim3((unsigned)((n + 255) / 256)), dim3(256), 0, stream>>>(src, dst, n);
  };
  cvt(x, xh, nX);
  cvt(wq, wqh, nWQ);
  cvt(wkv_a, wkvah, nWA);
  cvt(wkv_b, wkvbh, nWB);
  cvt(wo, woh, nWO);

  // q = x @ wq.T  -> f16 [T][3072]
  gemm_f16_wmma<<<dim3(T_SEQ / 128, QSTRIDE / 128), dim3(256), 0, stream>>>(
      xh, wqh, nullptr, qbuf, T_SEQ, QSTRIDE, D_MODEL);
  // RoPE on q_pe
  rope_q_kernel<<<dim3((T_SEQ * H_HEADS * 32) / 256), dim3(256), 0, stream>>>(qbuf);

  // kv_full = x @ wkv_a.T -> f32 [T][576]
  gemm_f16_wmma<<<dim3(T_SEQ / 128, (KVR + DR + 127) / 128), dim3(256), 0, stream>>>(
      xh, wkvah, kvfull, nullptr, T_SEQ, KVR + DR, D_MODEL);
  // layernorm latent + RoPE k_pe
  ln_ropek_kernel<<<dim3(T_SEQ), dim3(256), 0, stream>>>(kvfull, kv_gamma, kv_beta, kvlnh, kph);

  // kvb = kv_ln @ wkv_b.T -> f16 [T][4096]
  gemm_f16_wmma<<<dim3(T_SEQ / 128, KVBSTR / 128), dim3(256), 0, stream>>>(
      kvlnh, wkvbh, nullptr, kvbh, T_SEQ, KVBSTR, KVR);
  // V transpose for contiguous B-fragments in P*V
  transpose_v_kernel<<<dim3((H_HEADS * DV * T_SEQ) / 256), dim3(256), 0, stream>>>(kvbh, vTh);

  // flash attention
  mla_attn_kernel<<<dim3(T_SEQ / 128, H_HEADS), dim3(256), 0, stream>>>(
      qbuf, kvbh, kph, vTh, attnh);

  // out = attn @ wo.T -> f32 [T][2048]
  gemm_f16_wmma<<<dim3(T_SEQ / 128, OSTRIDE / 128), dim3(256), 0, stream>>>(
      attnh, woh, out, nullptr, T_SEQ, OSTRIDE, D_MODEL);
}
